// MLSTM_fixD_66726611911227
// MI455X (gfx1250) — matrix operand order repaired and checked by reading, not compiled, not run on hardware
//
#include <hip/hip_runtime.h>
#include <math.h>

// Problem dims (from reference)
#define T_   1024
#define B_   256
#define I_   256
#define H_   512
#define O_   256
#define KLAG 64
#define G3H  1536   // 3*H
#define KCAT 768    // I + H
#define BH   (B_*H_)

typedef __bf16 bf16;
typedef __attribute__((ext_vector_type(16))) __bf16 v16bf;
typedef __attribute__((ext_vector_type(8)))  __bf16 v8bf;
typedef __attribute__((ext_vector_type(8)))  float  v8f;

// fp32 -> bf16, round-to-nearest-even
__device__ __forceinline__ bf16 f2bf(float f) {
    unsigned u = __builtin_bit_cast(unsigned, f);
    unsigned r = (u + 0x7FFFu + ((u >> 16) & 1u)) >> 16;
    unsigned short s = (unsigned short)r;
    return __builtin_bit_cast(bf16, s);
}

__device__ __forceinline__ v16bf cat16(v8bf a, v8bf b) {
    return __builtin_shufflevector(a, b, 0,1,2,3,4,5,6,7,8,9,10,11,12,13,14,15);
}

__device__ __forceinline__ float sigmoidf_(float x) {
    return 1.0f / (1.0f + __expf(-x));
}

#define LDP 40  // padded LDS stride (bf16 elems); 80B rows, 16B aligned

// ---------------------------------------------------------------------------
// Fused gates GEMM + cell update.
// gates[m, g*H + h] = sum_k [x_t | h_{t-1}][m,k] * Wcat[g*H + h, k]  (+biases)
// Block tile: M=64 rows x 64 h-cols, all three gate slices, so the cell
// update (sigmoid/tanh + K=64 lag conv + ring write) runs in the epilogue.
// 8 waves as 4(M)x2(N): wave = 16 rows x 32 h-cols x 3 gates:
//   acc[3][2] (48 VGPRs) + bfr[6] (48) + af (8) ~ 110 live VGPRs,
// so the compiler can hoist all 12 ds_loads and stagger s_wait_dscnt.
// Grid: (H/64, B/64) = (8, 4) = 32 blocks.
// ---------------------------------------------------------------------------
__global__ __launch_bounds__(256)
void k_gates_cell(const bf16* __restrict__ Xt,        // (B, I) bf16, t-th slab
                  const bf16* __restrict__ hbf_in,    // (B, H) bf16
                  const bf16* __restrict__ Wcat,      // (3H, KCAT) bf16
                  const float* __restrict__ b_ih, const float* __restrict__ b_hh,
                  const float* __restrict__ wd,       // (KLAG, H): lag j -> wd[(j-1)*H+h]
                  float* __restrict__ ring,           // (KLAG, B, H)
                  float* __restrict__ hstate,         // (B, H) fp32
                  bf16* __restrict__ hbf_out,         // (B, H) bf16 (same buf as hbf_in)
                  int t)
{
    __shared__ bf16 sA[64 * LDP];    // A tile: 64 rows x 32 k
    __shared__ bf16 sB[192 * LDP];   // B tiles: 3 gates x 64 rows x 32 k

    const int tid  = threadIdx.x;
    const int lane = tid & 31;
    const int wave = tid >> 5;
    const int wm   = wave & 3;       // 0..3 -> 16 M-rows each
    const int wn   = wave >> 2;      // 0..1 -> 32 h-cols each
    const int nl   = lane & 15;
    const int half = lane >> 4;
    const int rowBase = blockIdx.y * 64;
    const int colBase = blockIdx.x * 64;   // h-column base

    v8f acc[3][2];
    #pragma unroll
    for (int g = 0; g < 3; ++g)
        #pragma unroll
        for (int j = 0; j < 2; ++j)
            #pragma unroll
            for (int r = 0; r < 8; ++r) acc[g][j][r] = 0.0f;

    for (int k0 = 0; k0 < KCAT; k0 += 32) {
        // Stage A (256 chunks of 8 bf16) + B (768 chunks) = 1024 chunks
        #pragma unroll
        for (int it = 0; it < 4; ++it) {
            int c = tid + it * 256;
            if (c < 256) {
                int row = c >> 2, kc = c & 3;
                int gk = k0 + kc * 8;
                const bf16* src = (gk < I_)
                    ? (Xt     + (size_t)(rowBase + row) * I_ + gk)
                    : (hbf_in + (size_t)(rowBase + row) * H_ + (gk - I_));
                *(v8bf*)&sA[row * LDP + kc * 8] = *(const v8bf*)src;
            } else {
                int cc = c - 256;
                int row = cc >> 2, kc = cc & 3;
                int gate = row >> 6, hrow = row & 63;
                int gk = k0 + kc * 8;
                const bf16* src = Wcat + (size_t)(gate * H_ + colBase + hrow) * KCAT + gk;
                *(v8bf*)&sB[row * LDP + kc * 8] = *(const v8bf*)src;
            }
        }
        __syncthreads();

        // A fragment (16x32): lane half k-interleave h*8+0..7, 16+h*8+0..7
        v16bf af;
        {
            const bf16* p = &sA[(wm * 16 + nl) * LDP + half * 8];
            v8bf c0 = *(const v8bf*)p;
            v8bf c1 = *(const v8bf*)(p + 16);
            af = cat16(c0, c1);
        }
        // All 6 B fragments up front (32x16 each: lane = N col, k = h*8+0..7, 16+h*8+0..7)
        v16bf bfr[6];
        #pragma unroll
        for (int g = 0; g < 3; ++g)
            #pragma unroll
            for (int j = 0; j < 2; ++j) {
                const bf16* p = &sB[(g * 64 + wn * 32 + j * 16 + nl) * LDP + half * 8];
                v8bf c0 = *(const v8bf*)p;
                v8bf c1 = *(const v8bf*)(p + 16);
                bfr[g * 2 + j] = cat16(c0, c1);
            }
        // WMMA burst
        #pragma unroll
        for (int g = 0; g < 3; ++g)
            #pragma unroll
            for (int j = 0; j < 2; ++j)
                acc[g][j] = __builtin_amdgcn_wmma_f32_16x16x32_bf16(
                    false, af, false, bfr[g * 2 + j], (short)0, acc[g][j], false, false);
        __syncthreads();
    }

    // Fused cell update. D layout: lane col = nl, rows = r + 8*half.
    const int slot = t & (KLAG - 1);
    #pragma unroll
    for (int j = 0; j < 2; ++j) {
        int h = colBase + wn * 32 + j * 16 + nl;
        float bi = b_ih[h]          + b_hh[h];
        float bg = b_ih[h + H_]     + b_hh[h + H_];
        float bo = b_ih[h + 2 * H_] + b_hh[h + 2 * H_];
        #pragma unroll
        for (int r = 0; r < 8; ++r) {
            int m   = rowBase + wm * 16 + half * 8 + r;
            int off = m * H_ + h;
            float gi = sigmoidf_(acc[0][j][r] + bi);
            float gg = tanhf(acc[1][j][r] + bg);
            float go = sigmoidf_(acc[2][j][r] + bo);
            float lag = 0.0f;
            #pragma unroll 4
            for (int s = 0; s < KLAG; ++s) {
                int jl = (t - s) & (KLAG - 1);   // lag of slot s; 0 means KLAG
                if (jl == 0) jl = KLAG;
                lag = fmaf(wd[(jl - 1) * H_ + h], ring[(size_t)s * BH + off], lag);
            }
            float cv = gi * gg - lag;
            ring[(size_t)slot * BH + off] = cv;  // overwrite oldest (own element)
            float hn = go * tanhf(cv);
            hstate[off] = hn;
            hbf_out[off] = f2bf(hn);
        }
    }
}

// ---------------------------------------------------------------------------
// Plain WMMA GEMM for the output projection:
// C[M,N] = A[M,Kdim] * Bm[N,Kdim]^T + biasA.  Block tile 128x128, 8 waves
// as 4(M)x2(N), wave tile 32x64. Grid: (N/128, M/128).
// ---------------------------------------------------------------------------
__global__ __launch_bounds__(256)
void k_wmma_gemm(const bf16* __restrict__ A, int lda,
                 const bf16* __restrict__ Bm, int ldb,
                 const float* __restrict__ biasA,
                 float* __restrict__ C, int ldc, int Kdim)
{
    __shared__ bf16 sA[128 * LDP];
    __shared__ bf16 sB[128 * LDP];

    const int tid  = threadIdx.x;
    const int lane = tid & 31;
    const int wave = tid >> 5;
    const int wm   = wave >> 1;
    const int wn   = wave & 1;
    const int rowBase = blockIdx.y * 128;
    const int colBase = blockIdx.x * 128;
    const int nl   = lane & 15;
    const int half = lane >> 4;

    v8f acc[2][4];
    #pragma unroll
    for (int s = 0; s < 2; ++s)
        #pragma unroll
        for (int j = 0; j < 4; ++j)
            #pragma unroll
            for (int r = 0; r < 8; ++r) acc[s][j][r] = 0.0f;

    for (int k0 = 0; k0 < Kdim; k0 += 32) {
        #pragma unroll
        for (int it = 0; it < 2; ++it) {
            int c   = tid + it * 256;
            int row = c >> 2;
            int kc  = c & 3;
            int gk  = k0 + kc * 8;
            *(v8bf*)&sA[row * LDP + kc * 8] =
                *(const v8bf*)(A  + (size_t)(rowBase + row) * lda + gk);
            *(v8bf*)&sB[row * LDP + kc * 8] =
                *(const v8bf*)(Bm + (size_t)(colBase + row) * ldb + gk);
        }
        __syncthreads();

        v16bf af[2];
        #pragma unroll
        for (int s = 0; s < 2; ++s) {
            const bf16* p = &sA[(wm * 32 + s * 16 + nl) * LDP + half * 8];
            v8bf c0 = *(const v8bf*)p;
            v8bf c1 = *(const v8bf*)(p + 16);
            af[s] = cat16(c0, c1);
        }
        v16bf bfr[4];
        #pragma unroll
        for (int j = 0; j < 4; ++j) {
            const bf16* p = &sB[(wn * 64 + j * 16 + nl) * LDP + half * 8];
            v8bf c0 = *(const v8bf*)p;
            v8bf c1 = *(const v8bf*)(p + 16);
            bfr[j] = cat16(c0, c1);
        }
        #pragma unroll
        for (int s = 0; s < 2; ++s)
            #pragma unroll
            for (int j = 0; j < 4; ++j)
                acc[s][j] = __builtin_amdgcn_wmma_f32_16x16x32_bf16(
                    false, af[s], false, bfr[j], (short)0, acc[s][j], false, false);
        __syncthreads();
    }

    const int col0 = colBase + wn * 64;
    const int row0 = rowBase + wm * 32;
    #pragma unroll
    for (int s = 0; s < 2; ++s) {
        #pragma unroll
        for (int j = 0; j < 4; ++j) {
            int col = col0 + j * 16 + nl;
            float bsum = biasA[col];
            int mrow = row0 + s * 16 + half * 8;
            #pragma unroll
            for (int r = 0; r < 8; ++r)
                C[(size_t)(mrow + r) * ldc + col] = acc[s][j][r] + bsum;
        }
    }
}

// ---------------------------------------------------------------------------
// Setup kernels
// ---------------------------------------------------------------------------
__global__ void k_cvt_bf16(const float* __restrict__ src, bf16* __restrict__ dst, int n) {
    int i = blockIdx.x * blockDim.x + threadIdx.x;
    int stride = gridDim.x * blockDim.x;
    for (; i < n; i += stride) dst[i] = f2bf(src[i]);
}

__global__ void k_wcat(const float* __restrict__ wih, const float* __restrict__ whh,
                       bf16* __restrict__ wcat) {
    int i = blockIdx.x * blockDim.x + threadIdx.x;
    int stride = gridDim.x * blockDim.x;
    const int n = G3H * KCAT;
    for (; i < n; i += stride) {
        int r = i / KCAT, k = i - r * KCAT;
        float v = (k < I_) ? wih[r * I_ + k] : whh[r * H_ + (k - I_)];
        wcat[i] = f2bf(v);
    }
}

// w_j = prod_{i=0..j-1} (i - d)/(i + 1), stored wd[(j-1)*H + h]
__global__ void k_wd(const float* __restrict__ bd, float* __restrict__ wd) {
    int h = blockIdx.x * blockDim.x + threadIdx.x;
    if (h >= H_) return;
    float d = 0.5f * sigmoidf_(bd[h]);
    float w = 1.0f;
    for (int j = 1; j <= KLAG; ++j) {
        w *= ((float)(j - 1) - d) / (float)j;
        wd[(j - 1) * H_ + h] = w;
    }
}

__global__ void k_init(float* __restrict__ ring, float* __restrict__ hstate,
                       bf16* __restrict__ hbf) {
    int i = blockIdx.x * blockDim.x + threadIdx.x;
    int stride = gridDim.x * blockDim.x;
    const int n = KLAG * BH;
    for (int k = i; k < n; k += stride) ring[k] = 0.0f;
    for (int k = i; k < BH; k += stride) { hstate[k] = 0.0f; hbf[k] = f2bf(0.0f); }
}

__global__ void k_finalize(const float* __restrict__ ring, const float* __restrict__ hstate,
                           float* __restrict__ out_hf, float* __restrict__ out_hc) {
    int i = blockIdx.x * blockDim.x + threadIdx.x;   // grid sized to KLAG*B*H
    if (i < BH) out_hf[i] = hstate[i];
    int k = i / BH, r = i - k * BH;
    int slot = (T_ - KLAG + k) & (KLAG - 1);         // hc[k] = c_{T-KLAG+k}
    out_hc[(size_t)k * BH + r] = ring[(size_t)slot * BH + r];
}

// ---------------------------------------------------------------------------
extern "C" void kernel_launch(void* const* d_in, const int* in_sizes, int n_in,
                              void* d_out, int out_size, void* d_ws, size_t ws_size,
                              hipStream_t stream)
{
    const float* inputs = (const float*)d_in[0];
    const float* bd     = (const float*)d_in[1];
    const float* w_ih   = (const float*)d_in[2];
    const float* w_hh   = (const float*)d_in[3];
    const float* b_ih   = (const float*)d_in[4];
    const float* b_hh   = (const float*)d_in[5];
    const float* w_out  = (const float*)d_in[6];
    const float* b_out  = (const float*)d_in[7];
    float* out = (float*)d_out;

    char* ws = (char*)d_ws;
    size_t off = 0;
    auto alloc = [&](size_t bytes) -> char* {
        char* p = ws + off;
        off += (bytes + 255) & ~(size_t)255;
        return p;
    };
    bf16*  Xbf    = (bf16*) alloc((size_t)T_ * B_ * I_ * sizeof(bf16)); // 134 MB
    bf16*  Wcat   = (bf16*) alloc((size_t)G3H * KCAT * sizeof(bf16));   // 2.4 MB
    bf16*  Woutbf = (bf16*) alloc((size_t)O_ * H_ * sizeof(bf16));      // 0.26 MB
    float* ring   = (float*)alloc((size_t)KLAG * BH * sizeof(float));   // 33.5 MB (L2)
    float* hstate = (float*)alloc((size_t)BH * sizeof(float));
    bf16*  hbf    = (bf16*) alloc((size_t)BH * sizeof(bf16));
    float* wd     = (float*)alloc((size_t)KLAG * H_ * sizeof(float));

    // Per-launch setup — deterministic, inside the captured graph
    k_cvt_bf16<<<2048, 256, 0, stream>>>(inputs, Xbf, T_ * B_ * I_);
    k_cvt_bf16<<<256, 256, 0, stream>>>(w_out, Woutbf, O_ * H_);
    k_wcat<<<1024, 256, 0, stream>>>(w_ih, w_hh, Wcat);
    k_wd<<<2, 256, 0, stream>>>(bd, wd);
    k_init<<<2048, 256, 0, stream>>>(ring, hstate, hbf);

    const dim3 blk(256);
    const dim3 gGates(H_ / 64, B_ / 64);    // (8, 4) = 32 blocks — fused gates + cell
    const dim3 gOut(O_ / 128, B_ / 128);    // (2, 2) — output projection

    for (int t = 0; t < T_; ++t) {
        const bf16* Xt = Xbf + (size_t)t * B_ * I_;
        k_gates_cell<<<gGates, blk, 0, stream>>>(Xt, hbf, Wcat, b_ih, b_hh,
                                                 wd, ring, hstate, hbf, t);
        k_wmma_gemm<<<gOut, blk, 0, stream>>>(hbf, H_, Woutbf, H_, b_out,
                                              out + (size_t)t * B_ * O_, O_, H_);
    }

    k_finalize<<<(KLAG * BH) / 256, blk, 0, stream>>>(
        ring, hstate,
        out + (size_t)T_ * B_ * O_,
        out + (size_t)T_ * B_ * O_ + BH);
}
